// BaseModernHopfield_87780541596164
// MI455X (gfx1250) — matrix operand order repaired and checked by reading, not compiled, not run on hardware
//
#include <hip/hip_runtime.h>
#include <hip/hip_bf16.h>

typedef __attribute__((ext_vector_type(16))) __bf16 v16bf;
typedef __attribute__((ext_vector_type(8)))  __bf16 v8bf;
typedef __attribute__((ext_vector_type(4)))  __bf16 v4bf;
typedef __attribute__((ext_vector_type(8)))  float  v8f;
typedef __attribute__((ext_vector_type(4)))  unsigned int u32x4;
typedef __attribute__((ext_vector_type(8)))  int i32x8;
typedef __attribute__((ext_vector_type(4)))  int i32x4;

#define BB 8
#define LL 1024
#define DD 512
#define HH 2048
#define KCHUNK 128          // K elements staged per TDM chunk (fwd)

union AFrag { v16bf v; v8bf h[2]; };

// ---------------------------------------------------------------------------
// TDM: stage a [n_rows x k_len] bf16 tile (row stride = `stride` elems) into
// LDS at byte offset lds_addr.  D# built per CDNA5 ISA §8 (groups 0/1).
// ---------------------------------------------------------------------------
__device__ __forceinline__ void tdm_load_tile(unsigned lds_addr, const __bf16* gptr,
                                              unsigned k_len, unsigned n_rows,
                                              unsigned stride) {
    unsigned long long ga = (unsigned long long)(size_t)gptr;
    u32x4 g0;
    g0[0] = 1u;                                            // count=1 (valid user D#)
    g0[1] = lds_addr;                                      // LDS byte address
    g0[2] = (unsigned)(ga & 0xffffffffull);                // global_addr[31:0]
    g0[3] = (unsigned)((ga >> 32) & 0x1ffffffull)
          | (2u << 30);                                    // global_addr[56:32] | type=2
    i32x8 g1;
    g1[0] = (int)(1u << 16);                               // wg_mask=0 | data_size=1 (2B)
    g1[1] = (int)((DD & 0xffffu) << 16);                   // tensor_dim0[15:0]=512
    g1[2] = (int)((HH & 0xffffu) << 16);                   // dim0_hi=0 | tensor_dim1[15:0]
    g1[3] = (int)((k_len & 0xffffu) << 16);                // dim1_hi=0 | tile_dim0
    g1[4] = (int)(n_rows & 0xffffu);                       // tile_dim1 | tile_dim2=0
    g1[5] = (int)stride;                                   // tensor_dim0_stride[31:0]
    g1[6] = 0;                                             // stride_hi | dim1_stride lo
    g1[7] = 0;
    i32x4 z4 = {};
#if __has_include(<hip/amd_detail/amd_gfx1250_TDM.h>)
    i32x8 z8 = {};
    __builtin_amdgcn_tensor_load_to_lds(g0, g1, z4, z4, z8, 0);   // 6-arg toolchain
#else
    __builtin_amdgcn_tensor_load_to_lds(g0, g1, z4, z4, 0);       // 5-arg ROCm 7.2
#endif
}

// ---------------------------------------------------------------------------
// Prep: split fp32 -> bf16 hi/lo pairs (x, and W in two layouts)
// ---------------------------------------------------------------------------
__global__ void prep_x_kernel(const float* __restrict__ x,
                              __bf16* __restrict__ xh, __bf16* __restrict__ xl) {
    int i = (blockIdx.x * blockDim.x + threadIdx.x) * 4;   // exact grid, no tail
    float4 v = *(const float4*)(x + i);
    float t[4] = {v.x, v.y, v.z, v.w};
    v4bf h, l;
#pragma unroll
    for (int j = 0; j < 4; ++j) {
        __bf16 hb = (__bf16)t[j];
        h[j] = hb;
        l[j] = (__bf16)(t[j] - (float)hb);
    }
    *(v4bf*)(xh + i) = h;
    *(v4bf*)(xl + i) = l;
}

__global__ void prep_w_kernel(const float* __restrict__ W,
                              __bf16* __restrict__ wrh, __bf16* __restrict__ wrl,
                              __bf16* __restrict__ wth, __bf16* __restrict__ wtl) {
    int i = blockIdx.x * blockDim.x + threadIdx.x;  // i over 512*2048, exact grid
    int d = i >> 11;
    int h = i & (HH - 1);
    float w = W[i];
    __bf16 hb = (__bf16)w;
    __bf16 lb = (__bf16)(w - (float)hb);
    wrh[i] = hb;                       // row-major [D,H]  (GEMM2 B operand)
    wrl[i] = lb;
    wth[(size_t)h * DD + d] = hb;      // transposed [H,D] (GEMM1 B operand)
    wtl[(size_t)h * DD + d] = lb;
}

// ---------------------------------------------------------------------------
// Kernel 1: block = 128(L) x 64(H) supertile, 8 waves in 4x2; each wave a 2x2
// array of 16x16 tiles.  B (W^T hi/lo) staged per-K-chunk into LDS by the TDM;
// A (x hi/lo) loaded direct (L2-resident).  bf16-split WMMA.  Elementwise
// relu/energy/grad in C fragments; Gs batch-reduced in registers.
// ---------------------------------------------------------------------------
__global__ void hopfield_fwd_kernel(const __bf16* __restrict__ xh, const __bf16* __restrict__ xl,
                                    const __bf16* __restrict__ wth, const __bf16* __restrict__ wtl,
                                    const float* __restrict__ beta_p,
                                    __bf16* __restrict__ gsh, __bf16* __restrict__ gsl,
                                    float* __restrict__ y_out) {
    __shared__ __bf16 sBh[64 * KCHUNK];      // 16 KB
    __shared__ __bf16 sBl[64 * KCHUNK];      // 16 KB

    const float beta = beta_p[0];
    const int lane = threadIdx.x & 31;
    const int wid  = threadIdx.x >> 5;
    const int wm   = wid & 3;                 // wave position in 4x2 grid
    const int wn   = wid >> 2;
    const int bm   = blockIdx.x >> 5;         // 8 M-blocks x 32 N-blocks
    const int bn   = blockIdx.x & 31;
    const int block_m = bm * 128;
    const int block_n = bn * 64;
    const int half = lane >> 4;
    const int l15  = lane & 15;
    const int offA = half * 8;                // A: K = half*8+j and 16+half*8+j
    const int offB = half * 8;                // B: K = half*8+j and 16+half*8+j

    const int mrow[2] = { block_m + wm * 32 + l15,
                          block_m + wm * 32 + 16 + l15 };
    const int ncol_l[2] = { wn * 32 + l15, wn * 32 + 16 + l15 };   // block-local B col

    const unsigned ldsB_h = (unsigned)(size_t)(__bf16*)sBh;   // low 32 bits = LDS offset
    const unsigned ldsB_l = (unsigned)(size_t)(__bf16*)sBl;

    float gacc[2][2][8];
#pragma unroll
    for (int mi = 0; mi < 2; ++mi)
#pragma unroll
        for (int ni = 0; ni < 2; ++ni)
#pragma unroll
            for (int i = 0; i < 8; ++i) gacc[mi][ni][i] = 0.f;
    float esum = 0.f;

    for (int b = 0; b < BB; ++b) {
        const __bf16* ah_row[2];
        const __bf16* al_row[2];
#pragma unroll
        for (int mi = 0; mi < 2; ++mi) {
            ah_row[mi] = xh + ((size_t)b * LL + mrow[mi]) * DD;
            al_row[mi] = xl + ((size_t)b * LL + mrow[mi]) * DD;
        }
        v8f acc[2][2];
#pragma unroll
        for (int mi = 0; mi < 2; ++mi)
#pragma unroll
            for (int ni = 0; ni < 2; ++ni) acc[mi][ni] = (v8f){};

        for (int ko = 0; ko < DD; ko += KCHUNK) {
            __syncthreads();                       // previous chunk fully consumed
            if (wid == 0) {
                // stage W^T hi/lo [64 cols x KCHUNK] for this block via TDM
                tdm_load_tile(ldsB_h, wth + (size_t)block_n * DD + ko, KCHUNK, 64, DD);
                tdm_load_tile(ldsB_l, wtl + (size_t)block_n * DD + ko, KCHUNK, 64, DD);
                __builtin_amdgcn_s_wait_tensorcnt(0);
            }
            __syncthreads();                       // chunk visible to all waves

            if (ko + KCHUNK < DD) {                // prefetch next A chunk into caches
                __builtin_prefetch(ah_row[0] + ko + KCHUNK, 0, 1);
                __builtin_prefetch(al_row[0] + ko + KCHUNK, 0, 1);
            }

#pragma unroll
            for (int k0 = 0; k0 < KCHUNK; k0 += 32) {
                AFrag ahi[2], alo[2], bhi[2], blo[2];
#pragma unroll
                for (int mi = 0; mi < 2; ++mi) {
                    ahi[mi].h[0] = *(const v8bf*)(ah_row[mi] + ko + k0 + offA);
                    ahi[mi].h[1] = *(const v8bf*)(ah_row[mi] + ko + k0 + 16 + offA);
                    alo[mi].h[0] = *(const v8bf*)(al_row[mi] + ko + k0 + offA);
                    alo[mi].h[1] = *(const v8bf*)(al_row[mi] + ko + k0 + 16 + offA);
                }
#pragma unroll
                for (int ni = 0; ni < 2; ++ni) {
                    const __bf16* sh = sBh + ncol_l[ni] * KCHUNK + k0 + offB;
                    const __bf16* sl = sBl + ncol_l[ni] * KCHUNK + k0 + offB;
                    bhi[ni].h[0] = *(const v8bf*)(sh);
                    bhi[ni].h[1] = *(const v8bf*)(sh + 16);
                    blo[ni].h[0] = *(const v8bf*)(sl);
                    blo[ni].h[1] = *(const v8bf*)(sl + 16);
                }
#pragma unroll
                for (int mi = 0; mi < 2; ++mi)
#pragma unroll
                    for (int ni = 0; ni < 2; ++ni) {
                        acc[mi][ni] = __builtin_amdgcn_wmma_f32_16x16x32_bf16(
                            false, ahi[mi].v, false, bhi[ni].v, (short)0, acc[mi][ni], false, false);
                        acc[mi][ni] = __builtin_amdgcn_wmma_f32_16x16x32_bf16(
                            false, ahi[mi].v, false, blo[ni].v, (short)0, acc[mi][ni], false, false);
                        acc[mi][ni] = __builtin_amdgcn_wmma_f32_16x16x32_bf16(
                            false, alo[mi].v, false, bhi[ni].v, (short)0, acc[mi][ni], false, false);
                    }
            }
        }
        // elementwise: e = relu(beta*z); energy & batch-reduced grad
#pragma unroll
        for (int mi = 0; mi < 2; ++mi)
#pragma unroll
            for (int ni = 0; ni < 2; ++ni)
#pragma unroll
                for (int i = 0; i < 8; ++i) {
                    float hv = beta * acc[mi][ni][i];
                    float e  = fmaxf(hv, 0.f);
                    esum += e * e;
                    gacc[mi][ni][i] -= beta * e;
                }
    }

    // write Gs tiles, pre-split to bf16 hi/lo (C layout: M = i + 8*half, N = lane&15)
#pragma unroll
    for (int mi = 0; mi < 2; ++mi)
#pragma unroll
        for (int ni = 0; ni < 2; ++ni)
#pragma unroll
            for (int i = 0; i < 8; ++i) {
                int lrow = block_m + wm * 32 + mi * 16 + i + 8 * half;
                int hcol = block_n + ncol_l[ni];
                size_t idx = (size_t)lrow * HH + hcol;
                float g = gacc[mi][ni][i];
                __bf16 hb = (__bf16)g;
                gsh[idx] = hb;
                gsl[idx] = (__bf16)(g - (float)hb);
            }

    // wave32 reduce of energy partial, one atomic per wave
    for (int off = 16; off > 0; off >>= 1) esum += __shfl_xor(esum, off, 32);
    if (lane == 0) atomicAdd(y_out, -0.5f * esum);
}

// ---------------------------------------------------------------------------
// Kernel 2: grad[l,d] = sum_h Gs[l,h] * W[d,h]  (Gs @ W^T), bf16-split WMMA,
// 2x2 register blocking per wave; block = 128(L) x 64(D).
// ---------------------------------------------------------------------------
__global__ void hopfield_bwd_kernel(const __bf16* __restrict__ gsh, const __bf16* __restrict__ gsl,
                                    const __bf16* __restrict__ wrh, const __bf16* __restrict__ wrl,
                                    float* __restrict__ grad) {
    const int lane = threadIdx.x & 31;
    const int wid  = threadIdx.x >> 5;
    const int wm   = wid & 3;
    const int wn   = wid >> 2;
    const int bm   = blockIdx.x >> 3;          // 8 M-blocks x 8 N-blocks
    const int bn   = blockIdx.x & 7;
    const int block_m = bm * 128;
    const int block_n = bn * 64;
    const int half = lane >> 4;
    const int l15  = lane & 15;
    const int offA = half * 8;
    const int offB = half * 8;

    const __bf16* ah_row[2];
    const __bf16* al_row[2];
    const __bf16* bh_row[2];
    const __bf16* bl_row[2];
#pragma unroll
    for (int i = 0; i < 2; ++i) {
        int m = block_m + wm * 32 + i * 16 + l15;
        int n = block_n + wn * 32 + i * 16 + l15;
        ah_row[i] = gsh + (size_t)m * HH;
        al_row[i] = gsl + (size_t)m * HH;
        bh_row[i] = wrh + (size_t)n * HH;      // W row d, contiguous in h=K
        bl_row[i] = wrl + (size_t)n * HH;
    }

    v8f acc[2][2];
#pragma unroll
    for (int mi = 0; mi < 2; ++mi)
#pragma unroll
        for (int ni = 0; ni < 2; ++ni) acc[mi][ni] = (v8f){};

#pragma unroll 2
    for (int k0 = 0; k0 < HH; k0 += 32) {
        AFrag ahi[2], alo[2], bhi[2], blo[2];
#pragma unroll
        for (int mi = 0; mi < 2; ++mi) {
            ahi[mi].h[0] = *(const v8bf*)(ah_row[mi] + k0 + offA);
            ahi[mi].h[1] = *(const v8bf*)(ah_row[mi] + k0 + 16 + offA);
            alo[mi].h[0] = *(const v8bf*)(al_row[mi] + k0 + offA);
            alo[mi].h[1] = *(const v8bf*)(al_row[mi] + k0 + 16 + offA);
        }
#pragma unroll
        for (int ni = 0; ni < 2; ++ni) {
            bhi[ni].h[0] = *(const v8bf*)(bh_row[ni] + k0 + offB);
            bhi[ni].h[1] = *(const v8bf*)(bh_row[ni] + k0 + offB + 16);
            blo[ni].h[0] = *(const v8bf*)(bl_row[ni] + k0 + offB);
            blo[ni].h[1] = *(const v8bf*)(bl_row[ni] + k0 + offB + 16);
        }
#pragma unroll
        for (int mi = 0; mi < 2; ++mi)
#pragma unroll
            for (int ni = 0; ni < 2; ++ni) {
                acc[mi][ni] = __builtin_amdgcn_wmma_f32_16x16x32_bf16(
                    false, ahi[mi].v, false, bhi[ni].v, (short)0, acc[mi][ni], false, false);
                acc[mi][ni] = __builtin_amdgcn_wmma_f32_16x16x32_bf16(
                    false, ahi[mi].v, false, blo[ni].v, (short)0, acc[mi][ni], false, false);
                acc[mi][ni] = __builtin_amdgcn_wmma_f32_16x16x32_bf16(
                    false, alo[mi].v, false, bhi[ni].v, (short)0, acc[mi][ni], false, false);
            }
    }

#pragma unroll
    for (int mi = 0; mi < 2; ++mi)
#pragma unroll
        for (int ni = 0; ni < 2; ++ni)
#pragma unroll
            for (int i = 0; i < 8; ++i) {
                int lrow = block_m + wm * 32 + mi * 16 + i + 8 * half;
                int dcol = block_n + wn * 32 + ni * 16 + l15;
                grad[(size_t)lrow * DD + dcol] = acc[mi][ni][i];
            }
}

// ---------------------------------------------------------------------------
extern "C" void kernel_launch(void* const* d_in, const int* in_sizes, int n_in,
                              void* d_out, int out_size, void* d_ws, size_t ws_size,
                              hipStream_t stream) {
    const float* x      = (const float*)d_in[0];   // [8,1024,512]
    const float* W      = (const float*)d_in[1];   // [512,2048]
    const float* beta_p = (const float*)d_in[2];   // [1]
    float* out = (float*)d_out;                    // [0]=y, [1..]=grad [1024,512]

    char* ws = (char*)d_ws;                        // 32 MB layout:
    __bf16* xh  = (__bf16*)(ws);                   //  0..8MB   x hi
    __bf16* xl  = (__bf16*)(ws + (8u  << 20));     //  8..16MB  x lo
    __bf16* wth = (__bf16*)(ws + (16u << 20));     // 16..18MB  W^T hi [H,D]
    __bf16* wtl = (__bf16*)(ws + (18u << 20));     // 18..20MB  W^T lo
    __bf16* wrh = (__bf16*)(ws + (20u << 20));     // 20..22MB  W hi   [D,H]
    __bf16* wrl = (__bf16*)(ws + (22u << 20));     // 22..24MB  W lo
    __bf16* gsh = (__bf16*)(ws + (24u << 20));     // 24..28MB  Gs hi  [L,H]
    __bf16* gsl = (__bf16*)(ws + (28u << 20));     // 28..32MB  Gs lo

    hipMemsetAsync(out, 0, sizeof(float), stream);              // y accumulator = 0

    prep_x_kernel<<<4096, 256, 0, stream>>>(x, xh, xl);         // 4,194,304 elems /4
    prep_w_kernel<<<4096, 256, 0, stream>>>(W, wrh, wrl, wth, wtl);

    hopfield_fwd_kernel<<<256, 256, 0, stream>>>(xh, xl, wth, wtl, beta_p,
                                                 gsh, gsl, out);
    hopfield_bwd_kernel<<<64, 256, 0, stream>>>(gsh, gsl, wrh, wrl, out + 1);
}
